// Instance_GAT_dgl_68487548502184
// MI455X (gfx1250) — matrix operand order repaired and checked by reading, not compiled, not run on hardware
//
#include <hip/hip_runtime.h>
#include <hip/hip_bf16.h>

// ---------------------------------------------------------------------------
// GAT (2-layer, DGL GATConv semantics) for MI455X / gfx1250, wave32.
//   Layer1: feat1 = emb@W1 (bf16 WMMA), edge-softmax over dst, aggregate.
//   Layer2: feat2 = relu(rst1+b1)@W2 (bf16 WMMA), edge-softmax, aggregate
//           directly into d_out (pre-initialized with b2 broadcast).
// All scatter math uses native u32/f32 atomics; whole working set (~66MB)
// is L2-resident on MI455X (192MB L2).
// ---------------------------------------------------------------------------

typedef __bf16 bf16_t;
typedef __attribute__((ext_vector_type(16))) __bf16 v16bf;
typedef __attribute__((ext_vector_type(8)))  __bf16 v8bf;
typedef __attribute__((ext_vector_type(8)))  float  v8f;
typedef __attribute__((ext_vector_type(4)))  float  v4f;

#define N_NODES 10000
#define IN_DIM  256
#define HID     128
#define HEADS   4
#define OUT_DIM 128
#define NEG_SLOPE 0.2f

// order-preserving float -> u32 map so segment-max uses global_atomic_max_u32
__device__ __forceinline__ unsigned f2ord(float f) {
  unsigned b = __float_as_uint(f);
  return (b & 0x80000000u) ? ~b : (b | 0x80000000u);
}
__device__ __forceinline__ float ord2f(unsigned u) {
  unsigned b = (u & 0x80000000u) ? (u & 0x7FFFFFFFu) : ~u;
  return __uint_as_float(b);
}

// --------------------------- small utility kernels -------------------------

__global__ void k_f32_to_bf16(const float* __restrict__ in,
                              bf16_t* __restrict__ out, int n) {
  int i = blockIdx.x * blockDim.x + threadIdx.x;
  if (i < n) out[i] = (bf16_t)in[i];
}

// in [R,C] f32  ->  out [C,R] bf16   (weights: row-major B -> B^T, K-contig)
__global__ void k_transpose_to_bf16(const float* __restrict__ in,
                                    bf16_t* __restrict__ out, int R, int C) {
  int i = blockIdx.x * blockDim.x + threadIdx.x;
  if (i < R * C) {
    int r = i / C, c = i - r * C;
    out[(size_t)c * R + r] = (bf16_t)in[i];
  }
}

__global__ void k_fill_f32(float* __restrict__ p, float v, int n) {
  int i = blockIdx.x * blockDim.x + threadIdx.x;
  if (i < n) p[i] = v;
}
__global__ void k_fill_u32(unsigned* __restrict__ p, unsigned v, int n) {
  int i = blockIdx.x * blockDim.x + threadIdx.x;
  if (i < n) p[i] = v;
}
// out[n*D+d] = b[d]   (bias broadcast into the accumulation target)
__global__ void k_bias_bcast(float* __restrict__ out,
                             const float* __restrict__ b, int n, int D) {
  int i = blockIdx.x * blockDim.x + threadIdx.x;
  if (i < n * D) out[i] = b[i % D];
}
// h = relu(rst + b) -> bf16 (layer-2 GEMM input)
__global__ void k_bias_relu_bf16(const float* __restrict__ rst,
                                 const float* __restrict__ b,
                                 bf16_t* __restrict__ out, int n, int HD) {
  int i = blockIdx.x * blockDim.x + threadIdx.x;
  if (i < n * HD) {
    float v = rst[i] + b[i % HD];
    out[i] = (bf16_t)(v > 0.f ? v : 0.f);
  }
}

// ------------------------------ WMMA GEMM ----------------------------------
// C[M,N] f32 = A[M,K] bf16 (row major) * B[K,N], with B given as BT[N,K].
// One wave per 16x16 C tile; K-loop of v_wmma_f32_16x16x32_bf16.
// A per-lane layout (contiguous K, same order as B): lane L holds row M=L%16,
//   K in [g*16, g*16+16), g = L/16  -> two 16B loads.
// B per-lane layout: lane L holds col N=L%16, K in [g*16, g*16+16) -> 32B.
__global__ void __launch_bounds__(256)
k_gemm_bf16_wmma(const bf16_t* __restrict__ A, const bf16_t* __restrict__ BT,
                 float* __restrict__ C, int M, int N, int K) {
  const int lane = threadIdx.x & 31;
  const int wave = threadIdx.x >> 5;
  const int ntn  = N >> 4;
  const int tile = blockIdx.x * 8 + wave;
  if (tile >= (M >> 4) * ntn) return;   // wave-uniform: EXEC all-ones below
  const int mt = tile / ntn;
  const int nt = tile - mt * ntn;
  const int r = lane & 15;
  const int g = lane >> 4;

  const bf16_t* Ap = A  + (size_t)(mt * 16 + r) * K;
  const bf16_t* Bp = BT + (size_t)(nt * 16 + r) * K + g * 16;

  v8f c = {};
  for (int k0 = 0; k0 < K; k0 += 32) {
    v8bf alo = *(const v8bf*)(Ap + k0 + g * 16);
    v8bf ahi = *(const v8bf*)(Ap + k0 + g * 16 + 8);
    v16bf b  = *(const v16bf*)(Bp + k0);
    v16bf a;
#pragma unroll
    for (int i = 0; i < 8; ++i) { a[i] = alo[i]; a[i + 8] = ahi[i]; }
    // (neg_a, A, neg_b, B, c_mod, C, reuse_a, reuse_b)
    c = __builtin_amdgcn_wmma_f32_16x16x32_bf16(false, a, false, b,
                                                (short)0, c, false, false);
  }
  // C/D layout: VGPR v, lanes 0-15 -> M=v, lanes 16-31 -> M=v+8; N=lane%16
  float* Cp = C + (size_t)(mt * 16 + g * 8) * N + nt * 16 + r;
#pragma unroll
  for (int v = 0; v < 8; ++v) Cp[(size_t)v * N] = c[v];
}

// -------------------------- attention scores -------------------------------
// el[n,h] = <feat[n,h,:], al[h,:]>, er likewise. One wave per (node,head),
// D==128 -> each lane owns 4 contiguous floats, wave-reduce via shfl_xor.
__global__ void k_attn_scores(const float* __restrict__ feat,
                              const float* __restrict__ al,
                              const float* __restrict__ ar,
                              float* __restrict__ el, float* __restrict__ er,
                              int H, int D) {
  const int n = blockIdx.x;
  const int h = threadIdx.x >> 5;
  const int lane = threadIdx.x & 31;
  const float* f = feat + (size_t)n * H * D + h * D + lane * 4;
  const float* L = al + h * D + lane * 4;
  const float* R = ar + h * D + lane * 4;
  v4f fv = *(const v4f*)f, lv = *(const v4f*)L, rv = *(const v4f*)R;
  float sl = 0.f, sr = 0.f;
#pragma unroll
  for (int i = 0; i < 4; ++i) { sl += fv[i] * lv[i]; sr += fv[i] * rv[i]; }
  for (int off = 16; off > 0; off >>= 1) {
    sl += __shfl_xor(sl, off, 32);
    sr += __shfl_xor(sr, off, 32);
  }
  if (lane == 0) { el[n * H + h] = sl; er[n * H + h] = sr; }
}

// ------------------------------ edge passes --------------------------------
// pass1: e = leaky_relu(el[src]+er[dst]); store e; atomic-max into m[dst].
__global__ void k_edge_pass1(const int* __restrict__ src,
                             const int* __restrict__ dst,
                             const float* __restrict__ el,
                             const float* __restrict__ er,
                             float* __restrict__ ebuf,
                             unsigned* __restrict__ m, int E, int H) {
  int t = blockIdx.x * blockDim.x + threadIdx.x;
  if (t >= E * H) return;
  int eid = t / H, h = t - eid * H;
  float e = el[src[eid] * H + h] + er[dst[eid] * H + h];
  e = e > 0.f ? e : NEG_SLOPE * e;
  ebuf[t] = e;
  atomicMax(&m[dst[eid] * H + h], f2ord(e));
}

// pass2: ex = exp(e - m[dst]); store ex; atomic-add into s[dst].
__global__ void k_edge_pass2(const int* __restrict__ dst,
                             float* __restrict__ ebuf,
                             const unsigned* __restrict__ m,
                             float* __restrict__ s, int E, int H) {
  int t = blockIdx.x * blockDim.x + threadIdx.x;
  if (t >= E * H) return;
  int eid = t / H, h = t - eid * H;
  float mv = ord2f(m[dst[eid] * H + h]);
  float ex = __expf(ebuf[t] - mv);
  ebuf[t] = ex;
  atomicAdd(&s[dst[eid] * H + h], ex);
}

// pass3: rst[dst,h,:] += (ex/s[dst,h]) * feat[src,h,:]. One wave per
// (edge,head); D==128 -> 4 floats/lane (float4 gather + 4 f32 atomic adds,
// all L2-resident on MI455X).
__global__ void k_edge_aggregate(const int* __restrict__ src,
                                 const int* __restrict__ dst,
                                 const float* __restrict__ ebuf,
                                 const float* __restrict__ s,
                                 const float* __restrict__ feat,
                                 float* __restrict__ rst, int E, int H, int D) {
  int w = (int)((blockIdx.x * (size_t)blockDim.x + threadIdx.x) >> 5);
  int lane = threadIdx.x & 31;
  if (w >= E * H) return;
  int eid = w / H, h = w - eid * H;
  float sv = s[dst[eid] * H + h];
  sv = sv > 0.f ? sv : 1.f;
  float a = ebuf[w] / sv;
  const float* fp = feat + (size_t)src[eid] * H * D + h * D + lane * 4;
  float* rp       = rst  + (size_t)dst[eid] * H * D + h * D + lane * 4;
  v4f fv = *(const v4f*)fp;
#pragma unroll
  for (int i = 0; i < 4; ++i) atomicAdd(&rp[i], a * fv[i]);
}

// ------------------------------- launcher ----------------------------------

extern "C" void kernel_launch(void* const* d_in, const int* in_sizes, int n_in,
                              void* d_out, int out_size, void* d_ws, size_t ws_size,
                              hipStream_t stream) {
  const float* emb = (const float*)d_in[0];
  const float* W1  = (const float*)d_in[1];
  const float* b1  = (const float*)d_in[2];
  const float* al1 = (const float*)d_in[3];
  const float* ar1 = (const float*)d_in[4];
  const float* W2  = (const float*)d_in[5];
  const float* b2  = (const float*)d_in[6];
  const float* al2 = (const float*)d_in[7];
  const float* ar2 = (const float*)d_in[8];
  const int*   src = (const int*)d_in[9];
  const int*   dst = (const int*)d_in[10];
  const int E = in_sizes[9];
  float* out = (float*)d_out;

  // workspace carve-up (256B aligned slices)
  char* p = (char*)d_ws;
  auto alloc = [&](size_t bytes) -> void* {
    void* r = (void*)p;
    p += (bytes + 255) & ~(size_t)255;
    return r;
  };
  bf16_t*  embb  = (bf16_t*)alloc((size_t)N_NODES * IN_DIM * 2);
  bf16_t*  w1t   = (bf16_t*)alloc((size_t)(HEADS * HID) * IN_DIM * 2);
  bf16_t*  w2t   = (bf16_t*)alloc((size_t)OUT_DIM * (HEADS * HID) * 2);
  float*   feat1 = (float*) alloc((size_t)N_NODES * HEADS * HID * 4);
  float*   rst1  = (float*) alloc((size_t)N_NODES * HEADS * HID * 4);
  bf16_t*  hbf   = (bf16_t*)alloc((size_t)N_NODES * HEADS * HID * 2);
  float*   feat2 = (float*) alloc((size_t)N_NODES * OUT_DIM * 4);
  float*   el1   = (float*) alloc((size_t)N_NODES * HEADS * 4);
  float*   er1   = (float*) alloc((size_t)N_NODES * HEADS * 4);
  unsigned* m1   = (unsigned*)alloc((size_t)N_NODES * HEADS * 4);
  float*   s1    = (float*) alloc((size_t)N_NODES * HEADS * 4);
  float*   el2   = (float*) alloc((size_t)N_NODES * 4);
  float*   er2   = (float*) alloc((size_t)N_NODES * 4);
  unsigned* m2   = (unsigned*)alloc((size_t)N_NODES * 4);
  float*   s2    = (float*) alloc((size_t)N_NODES * 4);
  float*   eb1   = (float*) alloc((size_t)E * HEADS * 4);
  float*   eb2   = (float*) alloc((size_t)E * 4);

  auto cdiv = [](long a, long b) { return (unsigned)((a + b - 1) / b); };

  // ---- precision conversion (one streaming pass) ----
  k_f32_to_bf16<<<cdiv((long)N_NODES * IN_DIM, 256), 256, 0, stream>>>(
      emb, embb, N_NODES * IN_DIM);
  k_transpose_to_bf16<<<cdiv((long)IN_DIM * HEADS * HID, 256), 256, 0, stream>>>(
      W1, w1t, IN_DIM, HEADS * HID);
  k_transpose_to_bf16<<<cdiv((long)HEADS * HID * OUT_DIM, 256), 256, 0, stream>>>(
      W2, w2t, HEADS * HID, OUT_DIM);

  // ---- layer 1 ----
  {
    int M = N_NODES, N = HEADS * HID, K = IN_DIM;         // 10000 x 512 x 256
    unsigned tiles = (unsigned)((M >> 4) * (N >> 4));     // 20000 -> 2500 blks
    k_gemm_bf16_wmma<<<cdiv(tiles, 8), 256, 0, stream>>>(embb, w1t, feat1, M, N, K);
  }
  k_attn_scores<<<N_NODES, HEADS * 32, 0, stream>>>(feat1, al1, ar1, el1, er1,
                                                    HEADS, HID);
  k_fill_u32<<<cdiv((long)N_NODES * HEADS, 256), 256, 0, stream>>>(m1, 0u, N_NODES * HEADS);
  k_fill_f32<<<cdiv((long)N_NODES * HEADS, 256), 256, 0, stream>>>(s1, 0.f, N_NODES * HEADS);
  k_fill_f32<<<cdiv((long)N_NODES * HEADS * HID, 256), 256, 0, stream>>>(
      rst1, 0.f, N_NODES * HEADS * HID);
  k_edge_pass1<<<cdiv((long)E * HEADS, 256), 256, 0, stream>>>(src, dst, el1, er1,
                                                               eb1, m1, E, HEADS);
  k_edge_pass2<<<cdiv((long)E * HEADS, 256), 256, 0, stream>>>(dst, eb1, m1, s1,
                                                               E, HEADS);
  k_edge_aggregate<<<cdiv((long)E * HEADS, 8), 256, 0, stream>>>(
      src, dst, eb1, s1, feat1, rst1, E, HEADS, HID);
  k_bias_relu_bf16<<<cdiv((long)N_NODES * HEADS * HID, 256), 256, 0, stream>>>(
      rst1, b1, hbf, N_NODES, HEADS * HID);

  // ---- layer 2 ----
  {
    int M = N_NODES, N = OUT_DIM, K = HEADS * HID;        // 10000 x 128 x 512
    unsigned tiles = (unsigned)((M >> 4) * (N >> 4));     // 5000 -> 625 blks
    k_gemm_bf16_wmma<<<cdiv(tiles, 8), 256, 0, stream>>>(hbf, w2t, feat2, M, N, K);
  }
  k_attn_scores<<<N_NODES, 32, 0, stream>>>(feat2, al2, ar2, el2, er2, 1, OUT_DIM);
  k_fill_u32<<<cdiv((long)N_NODES, 256), 256, 0, stream>>>(m2, 0u, N_NODES);
  k_fill_f32<<<cdiv((long)N_NODES, 256), 256, 0, stream>>>(s2, 0.f, N_NODES);
  k_bias_bcast<<<cdiv((long)N_NODES * OUT_DIM, 256), 256, 0, stream>>>(
      out, b2, N_NODES, OUT_DIM);                          // out starts at +b2
  k_edge_pass1<<<cdiv((long)E, 256), 256, 0, stream>>>(src, dst, el2, er2,
                                                       eb2, m2, E, 1);
  k_edge_pass2<<<cdiv((long)E, 256), 256, 0, stream>>>(dst, eb2, m2, s2, E, 1);
  k_edge_aggregate<<<cdiv((long)E, 8), 256, 0, stream>>>(
      src, dst, eb2, s2, feat2, out, E, 1, OUT_DIM);       // accumulate -> d_out
}
